// DeepFM_4131758539319
// MI455X (gfx1250) — matrix operand order repaired and checked by reading, not compiled, not run on hardware
//
#include <hip/hip_runtime.h>

// ---------------------------------------------------------------------------
// DeepFM forward on MI455X (gfx1250, wave32, WMMA).
// Dims (fixed): B=16384, F=26, V=100000, K=16, D=13, H=400, IN_DIM=429.
// GEMMs run in bf16 WMMA (v_wmma_f32_16x16x32_bf16) with f32 accumulation,
// software-pipelined (double-buffered fragments); sched_group_barrier pins
// the "prefetch k+1 loads before stage-k WMMAs" order so the backend cannot
// collapse the double buffer and serialize load->wait(0)->wmma.
// b1/b2 are omitted: a per-column constant cancels inside training-mode BN.
// ---------------------------------------------------------------------------

typedef __bf16 bf16_t;
typedef bf16_t v16bf __attribute__((ext_vector_type(16)));
typedef bf16_t v8bf  __attribute__((ext_vector_type(8)));
typedef float  v8f   __attribute__((ext_vector_type(8)));

#define NB    16384          // batch
#define NF    26             // fields
#define NV    100000         // vocab
#define NK    16             // emb dim
#define ND    13             // dense dim
#define NH    400            // hidden
#define LDX   448            // 429 padded to 14*32 (bf16 elems)
#define LDH   416            // 400 padded to 13*32
#define MT    (NB/16)        // 1024 m-tiles
#define EPSBN 1e-5f

#if __has_builtin(__builtin_amdgcn_sched_group_barrier)
#define SCHED_VMEM_READS(n) __builtin_amdgcn_sched_group_barrier(0x020, (n), 0)
#define SCHED_WMMAS(n)      __builtin_amdgcn_sched_group_barrier(0x008, (n), 0)
#else
#define SCHED_VMEM_READS(n)
#define SCHED_WMMAS(n)
#endif

__device__ __forceinline__ unsigned short f2bf(float x) {
    unsigned u = __float_as_uint(x);
    unsigned r = (u + 0x7FFFu + ((u >> 16) & 1u)) >> 16;   // round-nearest-even
    return (unsigned short)r;
}

// --- 0: transpose + pad + bf16-convert a weight matrix W[Kdim][400] -> Wt[400][ldt]
__global__ void prep_w_kernel(const float* __restrict__ W, unsigned short* __restrict__ Wt,
                              int Kdim, int ldt, int ntot) {
    int e = blockIdx.x * blockDim.x + threadIdx.x;
    if (e >= ntot) return;
    int n = e / ldt, k = e % ldt;
    float v = (k < Kdim) ? W[(size_t)k * NH + n] : 0.0f;
    Wt[e] = f2bf(v);
}

// --- 1: gather embeddings, FM 1st+2nd order, build bf16 X row.  One wave / row.
__global__ void build_kernel(const int* __restrict__ Xcat, const float* __restrict__ Xdense,
                             const float* __restrict__ fm1, const float* __restrict__ emb,
                             const float* __restrict__ Wd,  const float* __restrict__ bd,
                             unsigned short* __restrict__ Xb, float* __restrict__ y0) {
    int wave = (blockIdx.x * blockDim.x + threadIdx.x) >> 5;
    int lane = threadIdx.x & 31;
    if (wave >= NB) return;
    const int r    = wave;
    const int half = lane >> 4;      // which field of the pair
    const int k    = lane & 15;      // emb component
    unsigned short* xrow = Xb + (size_t)r * LDX;

    float ssum = 0.0f, ssq = 0.0f;   // per-k sums over fields (this half's fields)
    #pragma unroll
    for (int p = 0; p < 13; ++p) {
        int f   = 2 * p + half;
        int idx = Xcat[r * NF + f];
        float v = emb[((size_t)f * NV + idx) * NK + k];
        ssum += v;
        ssq  += v * v;
        xrow[f * NK + k] = f2bf(v);
    }
    // combine the two half-wave field groups -> totals per k (duplicated in halves)
    ssum += __shfl_xor(ssum, 16, 32);
    ssq  += __shfl_xor(ssq , 16, 32);
    float ix = ssum * ssum - ssq;
    // sum ix over k = 0..15 (halves hold identical copies)
    #pragma unroll
    for (int d = 1; d < 16; d <<= 1) ix += __shfl_xor(ix, d, 32);
    float fm_y = 0.5f * ix;

    // 1st order + dense linear; also write dense slice + zero pad into X
    float t = 0.0f;
    if (lane < NF) {
        int idx = Xcat[r * NF + lane];
        t = fm1[(size_t)lane * NV + idx];
    }
    if (lane < ND) {
        float xd = Xdense[r * ND + lane];
        t += xd * Wd[lane];
        xrow[NF * NK + lane] = f2bf(xd);
    } else {
        xrow[NF * NK + lane] = 0;    // lanes 13..31 zero cols 429..447
    }
    #pragma unroll
    for (int d = 1; d < 32; d <<= 1) t += __shfl_xor(t, d, 32);
    if (lane == 0) y0[r] = t + bd[0] + fm_y;
}

// --- 2: zero the stats accumulators
__global__ void zero_kernel(float* __restrict__ p, int n) {
    int i = blockIdx.x * blockDim.x + threadIdx.x;
    if (i < n) p[i] = 0.0f;
}

// --- 3: bf16 WMMA GEMM  Z[16384 x 400] = X[16384 x (32*KTILES)] @ Wt^T
//     Wt stored as [400][LD] (K contiguous).  One wave -> 16x80 output strip.
//     Fully unrolled + double-buffered fragments; per stage the scheduler is
//     pinned to "12 VMEM reads (prefetch kt+1), then 5 WMMAs (stage kt)".
template <int KTILES, int LD>
__global__ void __launch_bounds__(256)
gemm_bf16_kernel(const unsigned short* __restrict__ X,
                 const unsigned short* __restrict__ Wt,
                 float* __restrict__ Z) {
    int wave  = (blockIdx.x * blockDim.x + threadIdx.x) >> 5;
    int lane  = threadIdx.x & 31;
    int mtile = wave / 5;            // grid sized exactly: no partial waves
    int ng    = wave % 5;            // n-group: 5 tiles of 16 -> cols [ng*80, +80)
    if (mtile >= MT) return;         // wave-uniform; EXEC stays all-ones
    const int half   = lane >> 4;
    const int within = lane & 15;

    // A fragment source: row (mtile*16+within), this lane's 16 contiguous K values
    const unsigned short* xrow = X + (size_t)(mtile * 16 + within) * LD + half * 16;
    // B fragment sources: column n, this lane's 16 contiguous K values
    const unsigned short* wb[5];
    #pragma unroll
    for (int nt = 0; nt < 5; ++nt) {
        int n = (ng * 5 + nt) * 16 + within;
        wb[nt] = Wt + (size_t)n * LD + half * 16;
    }

    union ABu { v16bf v; v8bf h[2]; };
    ABu   a[2];
    v16bf b[2][5];
    v8f   acc[5] = {};

    // prologue: prefetch k-tile 0
    a[0].h[0] = *(const v8bf*)(const void*)(xrow);
    a[0].h[1] = *(const v8bf*)(const void*)(xrow + 8);
    #pragma unroll
    for (int nt = 0; nt < 5; ++nt)
        b[0][nt] = *(const v16bf*)(const void*)(wb[nt]);

    #pragma unroll
    for (int kt = 0; kt < KTILES; ++kt) {
        const int cur = kt & 1, nxt = cur ^ 1;
        if (kt + 1 < KTILES) {       // prefetch next k-tile into the other buffer
            a[nxt].h[0] = *(const v8bf*)(const void*)(xrow + (kt + 1) * 32);
            a[nxt].h[1] = *(const v8bf*)(const void*)(xrow + (kt + 1) * 32 + 8);
            #pragma unroll
            for (int nt = 0; nt < 5; ++nt)
                b[nxt][nt] = *(const v16bf*)(const void*)(wb[nt] + (kt + 1) * 32);
        }
        #pragma unroll
        for (int nt = 0; nt < 5; ++nt)
            acc[nt] = __builtin_amdgcn_wmma_f32_16x16x32_bf16(
                false, a[cur].v, false, b[cur][nt], (short)0, acc[nt], false, false);
        // pin schedule: prefetch loads must stay above this stage's WMMAs
        if (kt + 1 < KTILES) SCHED_VMEM_READS(12);
        SCHED_WMMAS(5);
    }

    #pragma unroll
    for (int nt = 0; nt < 5; ++nt) {
        int n = (ng * 5 + nt) * 16 + within;
        #pragma unroll
        for (int rr = 0; rr < 8; ++rr) {
            int row = mtile * 16 + half * 8 + rr;                    // D layout (ISA 7.12.2)
            Z[(size_t)row * NH + n] = acc[nt][rr];
        }
    }
}

// --- 4: per-column sum / sum-of-squares over the batch (LDS atomics -> global)
__global__ void colstats_kernel(const float* __restrict__ Z,
                                float* __restrict__ gsum, float* __restrict__ gsq) {
    __shared__ float ssum[NH];
    __shared__ float ssq[NH];
    for (int i = threadIdx.x; i < NH; i += blockDim.x) { ssum[i] = 0.0f; ssq[i] = 0.0f; }
    __syncthreads();
    size_t base  = (size_t)blockIdx.x * 16384;                       // elems per block
    size_t total = (size_t)NB * NH;
    for (int it = 0; it < 64; ++it) {
        size_t e = base + threadIdx.x + (size_t)it * 256;
        if (e < total) {
            float v = Z[e];
            int   c = (int)(e % NH);
            atomicAdd(&ssum[c], v);
            atomicAdd(&ssq[c], v * v);
        }
    }
    __syncthreads();
    for (int i = threadIdx.x; i < NH; i += blockDim.x) {
        atomicAdd(&gsum[i], ssum[i]);
        atomicAdd(&gsq[i],  ssq[i]);
    }
}

// --- 5: stats -> affine (scale, shift) per column
__global__ void finalize_stats_kernel(const float* __restrict__ sum, const float* __restrict__ sq,
                                      const float* __restrict__ g,   const float* __restrict__ be,
                                      float* __restrict__ scale, float* __restrict__ shift) {
    int c = blockIdx.x * blockDim.x + threadIdx.x;
    if (c >= NH) return;
    float m   = sum[c] * (1.0f / NB);
    float var = sq[c] * (1.0f / NB) - m * m;
    float inv = rsqrtf(var + EPSBN);
    float s   = g[c] * inv;
    scale[c] = s;
    shift[c] = be[c] - m * s;
}

// --- 6: BN + ReLU -> bf16 hidden activations, padded to LDH columns
__global__ void bnrelu_kernel(const float* __restrict__ Z,
                              const float* __restrict__ scale, const float* __restrict__ shift,
                              unsigned short* __restrict__ Hb) {
    size_t e = (size_t)blockIdx.x * blockDim.x + threadIdx.x;
    if (e >= (size_t)NB * LDH) return;
    int    c   = (int)(e % LDH);
    size_t row = e / LDH;
    float h = 0.0f;
    if (c < NH) h = fmaxf(scale[c] * Z[row * NH + c] + shift[c], 0.0f);
    Hb[e] = f2bf(h);
}

// --- 7: fused BN2 + ReLU + GEMV(W3) + combine.  One wave / row.
__global__ void final_kernel(const float* __restrict__ Z,
                             const float* __restrict__ scale, const float* __restrict__ shift,
                             const float* __restrict__ W3, const float* __restrict__ b3,
                             const float* __restrict__ y0, float* __restrict__ out) {
    int wave = (blockIdx.x * blockDim.x + threadIdx.x) >> 5;
    int lane = threadIdx.x & 31;
    if (wave >= NB) return;
    float acc = 0.0f;
    for (int c = lane; c < NH; c += 32) {
        float h = fmaxf(scale[c] * Z[(size_t)wave * NH + c] + shift[c], 0.0f);
        acc += h * W3[c];
    }
    #pragma unroll
    for (int d = 1; d < 32; d <<= 1) acc += __shfl_xor(acc, d, 32);
    if (lane == 0) out[wave] = y0[wave] + acc + b3[0];
}

// ---------------------------------------------------------------------------
extern "C" void kernel_launch(void* const* d_in, const int* in_sizes, int n_in,
                              void* d_out, int out_size, void* d_ws, size_t ws_size,
                              hipStream_t stream) {
    const int*   X_cat   = (const int*)  d_in[0];
    const float* X_dense = (const float*)d_in[1];
    const float* fm1_t   = (const float*)d_in[2];
    const float* emb_t   = (const float*)d_in[3];
    const float* Wd      = (const float*)d_in[4];
    const float* bd      = (const float*)d_in[5];
    const float* W1      = (const float*)d_in[6];
    // d_in[7] = b1 (no-op under batch-norm)
    const float* g1      = (const float*)d_in[8];
    const float* be1     = (const float*)d_in[9];
    const float* W2      = (const float*)d_in[10];
    // d_in[11] = b2 (no-op under batch-norm)
    const float* g2      = (const float*)d_in[12];
    const float* be2     = (const float*)d_in[13];
    const float* W3      = (const float*)d_in[14];
    const float* b3      = (const float*)d_in[15];
    float* out = (float*)d_out;

    // workspace layout (all offsets 256B-aligned)
    char* ws = (char*)d_ws;
    unsigned short* Xb   = (unsigned short*)(ws);                         // 16384*448*2 = 14,680,064
    unsigned short* W1t  = (unsigned short*)(ws + 14680064);              //   400*448*2 =    358,400
    unsigned short* W2t  = (unsigned short*)(ws + 15038464);              //   400*416*2 =    332,800
    float*          y0   = (float*)(ws + 15371264);                       //   16384*4   =     65,536
    float*          st   = (float*)(ws + 15436800);                       //   1600*4    =      6,400
    float*          ssh  = (float*)(ws + 15443200);                       //   1600*4    =      6,400
    float*          Z    = (float*)(ws + 15449600);                       // 16384*400*4 = 26,214,400
    unsigned short* Hb   = Xb;  // reuse: X no longer needed after GEMM1  // 16384*416*2 fits

    // 0) weight transpose/pad/convert
    prep_w_kernel<<<(NH * LDX + 255) / 256, 256, 0, stream>>>(W1, W1t, 429, LDX, NH * LDX);
    prep_w_kernel<<<(NH * LDH + 255) / 256, 256, 0, stream>>>(W2, W2t, NH,  LDH, NH * LDH);
    // 1) gathers + FM terms + bf16 X
    build_kernel<<<NB / 8, 256, 0, stream>>>(X_cat, X_dense, fm1_t, emb_t, Wd, bd, Xb, y0);
    // 2) clear BN accumulators
    zero_kernel<<<(1600 + 255) / 256, 256, 0, stream>>>(st, 1600);
    // 3) layer 1 GEMM (bf16 WMMA): 1024 m-tiles * 5 n-groups = 5120 waves
    gemm_bf16_kernel<14, LDX><<<(MT * 5) / 8, 256, 0, stream>>>(Xb, W1t, Z);
    // 4) BN1 stats
    colstats_kernel<<<(NB * NH) / 16384, 256, 0, stream>>>(Z, st, st + 400);
    finalize_stats_kernel<<<2, 256, 0, stream>>>(st, st + 400, g1, be1, ssh, ssh + 400);
    // 5) BN1 + ReLU -> bf16 H
    bnrelu_kernel<<<((size_t)NB * LDH) / 256, 256, 0, stream>>>(Z, ssh, ssh + 400, Hb);
    // 6) layer 2 GEMM (bf16 WMMA), Z reused as output
    gemm_bf16_kernel<13, LDH><<<(MT * 5) / 8, 256, 0, stream>>>(Hb, W2t, Z);
    // 7) BN2 stats
    colstats_kernel<<<(NB * NH) / 16384, 256, 0, stream>>>(Z, st + 800, st + 1200);
    finalize_stats_kernel<<<2, 256, 0, stream>>>(st + 800, st + 1200, g2, be2, ssh + 800, ssh + 1200);
    // 8) fused BN2+ReLU+GEMV+combine
    final_kernel<<<NB / 8, 256, 0, stream>>>(Z, ssh + 800, ssh + 1200, W3, b3, y0, out);
}
